// GraphTransformerLayer_20306605375672
// MI455X (gfx1250) — matrix operand order repaired and checked by reading, not compiled, not run on hardware
//
#include <hip/hip_runtime.h>
#include <hip/hip_bf16.h>

// ---------------------------------------------------------------------------
// Graph-transformer layer for MI455X (gfx1250, wave32, WMMA).
// - All GEMMs: v_wmma_f32_16x16x32_bf16, f32 accumulation.
// - B operands (weights, V) pre-swizzled into WMMA fragment order so every
//   fragment load is a contiguous 32B/lane global_load_b128 pair.
// - Flash attention fuses scale+bias+softmax; attn_bias (256MB) read once.
// ---------------------------------------------------------------------------

typedef __attribute__((ext_vector_type(16))) __bf16       v16bf;
typedef __attribute__((ext_vector_type(8)))  float        v8f;
typedef __attribute__((ext_vector_type(8)))  unsigned int v8u;

#define HD     1024
#define SEQ    1024
#define NHEADS 16
#define DHEAD  64
#define BATCH  4
#define ROWS   (BATCH * SEQ)   // 4096

__device__ inline unsigned short f2bf(float f) {
  unsigned u = __builtin_bit_cast(unsigned, f);
  u += 0x7fffu + ((u >> 16) & 1u);          // round-to-nearest-even
  return (unsigned short)(u >> 16);
}
__device__ inline float bf2f(unsigned short h) {
  return __builtin_bit_cast(float, ((unsigned)h) << 16);
}
__device__ inline v8f zero8() {
  v8f z = {0.f, 0.f, 0.f, 0.f, 0.f, 0.f, 0.f, 0.f};
  return z;
}
__device__ inline v8f wmma_bf(v16bf a, v16bf b, v8f c) {
  return __builtin_amdgcn_wmma_f32_16x16x32_bf16(false, a, false, b,
                                                 (short)0, c, false, false);
}

// --- A fragment (16x32, row-major src). k order used here (same as packed B): lane m=L%16, half=L/16;
//     K = 16*half + 2*v(+1) -> two contiguous 16B runs,
//     compiler coalesces into 2x global_load_b128.
__device__ inline v16bf load_a_g(const unsigned short* __restrict__ base,
                                 int ld, int m0, int k0, int lane) {
  const int m = lane & 15, hh = lane >> 4;
  const unsigned short* row = base + (size_t)(m0 + m) * ld + k0;
  v8u u;
#pragma unroll
  for (int v = 0; v < 8; ++v) {
    const int k = (hh << 4) + (v << 1);
    u[v] = *(const unsigned int*)(row + k);
  }
  return __builtin_bit_cast(v16bf, u);
}

// --- B fragment where B[k][n] = Src[n][k] (Src row-major): contiguous 32B
//     per lane; used for Q @ K^T with no transpose in memory.
__device__ inline v16bf load_bT_g(const unsigned short* __restrict__ base,
                                  int ld, int k0, int n0, int lane) {
  const int n = lane & 15, hh = lane >> 4;
  const unsigned short* row = base + (size_t)(n0 + n) * ld + k0 + (hh << 4);
  v8u u;
#pragma unroll
  for (int v = 0; v < 8; ++v) u[v] = *(const unsigned int*)(row + 2 * v);
  return __builtin_bit_cast(v16bf, u);
}

// --- A fragment from a 16x32 bf16 tile in LDS (row stride 32 shorts).
__device__ inline v16bf load_a_lds(const unsigned short* __restrict__ P, int lane) {
  const int m = lane & 15, hh = lane >> 4;
  const unsigned short* row = P + m * 32;
  v8u u;
#pragma unroll
  for (int v = 0; v < 8; ++v) {
    const int k = (hh << 4) + (v << 1);
    u[v] = *(const unsigned int*)(row + k);
  }
  return __builtin_bit_cast(v16bf, u);
}

// ---------------------------------------------------------------------------
// Pack an f32 KxN row-major weight into WMMA-B fragment order (bf16):
//   out[kt][nt][lane][e], kt=K/32, nt=N/16; lane: n=nt*16+(lane&15),
//   k = kt*32 + 16*(lane>>4) + e   (e in 0..15).
// A B-fragment load then becomes one contiguous 32B read per lane.
// ---------------------------------------------------------------------------
__global__ __launch_bounds__(256) void pack_b_kernel(
    const float* __restrict__ W, unsigned short* __restrict__ out,
    int N, int ntShift, int total) {
  const int idx = blockIdx.x * 256 + threadIdx.x;
  if (idx >= total) return;
  const int e = idx & 15, lane = (idx >> 4) & 31;
  const int rest = idx >> 9;
  const int nt = rest & ((1 << ntShift) - 1);
  const int kt = rest >> ntShift;
  const int k  = (kt << 5) + ((lane >> 4) << 4) + e;
  const int nn = (nt << 4) + (lane & 15);
  out[idx] = f2bf(W[(size_t)k * N + nn]);
}

// ---------------------------------------------------------------------------
// Repack V (bf16, (B*S, H) row-major, head-interleaved) into per-(b,h)
// WMMA-B fragment order: out[b][h][kvt(32)][nt(4)][lane(32)][e(16)].
// ---------------------------------------------------------------------------
__global__ __launch_bounds__(256) void pack_v_kernel(
    const unsigned short* __restrict__ Vsrc, unsigned short* __restrict__ out) {
  const int idx = blockIdx.x * 256 + threadIdx.x;   // total = ROWS*HD = 4M
  const int e    = idx & 15;
  const int lane = (idx >> 4) & 31;
  const int nt   = (idx >> 9) & 3;
  const int kvt  = (idx >> 11) & 31;
  const int h    = (idx >> 16) & 15;
  const int b    = idx >> 20;
  const int kv   = (kvt << 5) + ((lane >> 4) << 4) + e;
  const int col  = h * DHEAD + (nt << 4) + (lane & 15);
  out[idx] = Vsrc[((size_t)b * SEQ + kv) * HD + col];
}

// ---------------------------------------------------------------------------
// LayerNorm over last dim (H=1024), one block per row; optional residual-sum,
// f32 LN output, and bf16 LN output.
// ---------------------------------------------------------------------------
__global__ __launch_bounds__(256) void ln_kernel(
    const float* __restrict__ in, const float* __restrict__ res,
    const float* __restrict__ g, const float* __restrict__ bb,
    float* __restrict__ out_sum, float* __restrict__ out_ln,
    unsigned short* __restrict__ out_bf) {
  __shared__ float s1[256], s2[256];
  const int row = blockIdx.x, t = threadIdx.x;
  const float* p = in + (size_t)row * HD;
  float v[4];
  float a = 0.f, q = 0.f;
#pragma unroll
  for (int i = 0; i < 4; ++i) {
    v[i] = p[t + 256 * i];
    a += v[i];
    q += v[i] * v[i];
  }
  s1[t] = a;
  s2[t] = q;
  __syncthreads();
  for (int off = 128; off > 0; off >>= 1) {
    if (t < off) { s1[t] += s1[t + off]; s2[t] += s2[t + off]; }
    __syncthreads();
  }
  const float mean = s1[0] * (1.f / HD);
  const float var  = s2[0] * (1.f / HD) - mean * mean;
  const float rstd = rsqrtf(var + 1e-5f);
#pragma unroll
  for (int i = 0; i < 4; ++i) {
    const int c = t + 256 * i;
    const float y = (v[i] - mean) * rstd * g[c] + bb[c];
    const size_t idx = (size_t)row * HD + c;
    if (out_ln)  out_ln[idx] = y;
    if (out_sum) out_sum[idx] = res[idx] + y;
    if (out_bf)  out_bf[idx] = f2bf(y);
  }
}

// ---------------------------------------------------------------------------
// bf16 WMMA GEMM: C(MxN) = A(MxK, row-major bf16) @ Bp (pre-swizzled bf16).
// One wave per 32x64 tile: 2x4 fragments, 8 v_wmma per 32-wide K-step.
// tnShift = log2(N/64).
// ---------------------------------------------------------------------------
__global__ __launch_bounds__(128) void gemm_bf16_kernel(
    const unsigned short* __restrict__ A, const unsigned short* __restrict__ Bp,
    float* __restrict__ Cf, unsigned short* __restrict__ Cb,
    int M, int N, int K, int tnShift) {
  const int lane = threadIdx.x;
  const int tile = blockIdx.x * 4 + threadIdx.y;
  const int tm = (tile >> tnShift) << 5;
  const int tn = (tile & ((1 << tnShift) - 1)) << 6;
  if (tm >= M) return;  // uniform per wave -> EXEC all-ones for WMMA
  const int n = lane & 15, hh = lane >> 4;

  v8f acc[2][4];
#pragma unroll
  for (int mi = 0; mi < 2; ++mi)
#pragma unroll
    for (int ni = 0; ni < 4; ++ni) acc[mi][ni] = zero8();

  const unsigned short* bt = Bp + (((size_t)tn >> 4) << 9) + (lane << 4);
  const size_t bstep = ((size_t)(N >> 4)) << 9;  // one 32-K-slab of packed B
  for (int k0 = 0; k0 < K; k0 += 32) {
    const v16bf a0 = load_a_g(A, K, tm, k0, lane);
    const v16bf a1 = load_a_g(A, K, tm + 16, k0, lane);
    v16bf bf[4];
#pragma unroll
    for (int ni = 0; ni < 4; ++ni)
      bf[ni] = *(const v16bf*)(bt + ((size_t)ni << 9));
    if (k0 + 32 < K) __builtin_prefetch(bt + bstep, 0, 3);  // global_prefetch_b8
#pragma unroll
    for (int ni = 0; ni < 4; ++ni) {
      acc[0][ni] = wmma_bf(a0, bf[ni], acc[0][ni]);
      acc[1][ni] = wmma_bf(a1, bf[ni], acc[1][ni]);
    }
    bt += bstep;
  }
#pragma unroll
  for (int mi = 0; mi < 2; ++mi)
#pragma unroll
    for (int ni = 0; ni < 4; ++ni)
#pragma unroll
      for (int r = 0; r < 8; ++r) {
        const int rr = tm + mi * 16 + (hh << 3) + r;
        const int cc = tn + ni * 16 + n;
        if (Cf) Cf[(size_t)rr * N + cc] = acc[mi][ni][r];
        if (Cb) Cb[(size_t)rr * N + cc] = f2bf(acc[mi][ni][r]);
      }
}

// ---------------------------------------------------------------------------
// Flash attention: one wave per (b, head, 32-query tile). Streams 32-wide KV
// tiles; scores = (Q K^T)*scale + bias, online softmax, O += P @ V via WMMA.
// K read via contiguous B^T fragments; V pre-swizzled; P exchanged through a
// per-wave LDS tile (C-layout -> A-layout).
// ---------------------------------------------------------------------------
__global__ __launch_bounds__(128) void attn_kernel(
    const unsigned short* __restrict__ Qb, const unsigned short* __restrict__ Kb,
    const unsigned short* __restrict__ Vpk, const float* __restrict__ bias,
    float* __restrict__ Of, unsigned short* __restrict__ Ob) {
  __shared__ unsigned short Pl[4][32][32];  // per-wave P tile (bf16)
  const int lane = threadIdx.x, wave = threadIdx.y;
  const int tile = blockIdx.x * 4 + wave;   // B*NH*(S/32) = 2048 tiles
  const int qb = tile & 31;
  const int h  = (tile >> 5) & 15;
  const int b  = tile >> 9;
  const int q0 = qb << 5;
  const int n = lane & 15, hh = lane >> 4;

  const unsigned short* Qp = Qb + (size_t)b * SEQ * HD + h * DHEAD;
  const unsigned short* Kp = Kb + (size_t)b * SEQ * HD + h * DHEAD;
  const unsigned short* Vp =
      Vpk + (((size_t)(b * NHEADS + h) * 32 * 4) << 9) + (lane << 4);
  const float* Bp = bias + ((size_t)b * NHEADS + h) * SEQ * SEQ;

  // Q fragments (32x64) resident in VGPRs for the whole KV sweep.
  v16bf aq[2][2];
#pragma unroll
  for (int qi = 0; qi < 2; ++qi)
#pragma unroll
    for (int di = 0; di < 2; ++di)
      aq[qi][di] = load_a_g(Qp, HD, q0 + qi * 16, di * 32, lane);

  v8f o[2][4];
  float mrow[2][8], lrow[2][8];
#pragma unroll
  for (int qi = 0; qi < 2; ++qi) {
#pragma unroll
    for (int nt = 0; nt < 4; ++nt) o[qi][nt] = zero8();
#pragma unroll
    for (int r = 0; r < 8; ++r) { mrow[qi][r] = -1e30f; lrow[qi][r] = 0.f; }
  }

  for (int kv0 = 0; kv0 < SEQ; kv0 += 32) {
    // ---- S = Q @ K^T (f32 accum) ----
    v8f s[2][2];
#pragma unroll
    for (int qi = 0; qi < 2; ++qi)
#pragma unroll
      for (int ni = 0; ni < 2; ++ni) s[qi][ni] = zero8();
#pragma unroll
    for (int di = 0; di < 2; ++di) {
      const v16bf bk0 = load_bT_g(Kp, HD, di * 32, kv0, lane);
      const v16bf bk1 = load_bT_g(Kp, HD, di * 32, kv0 + 16, lane);
      s[0][0] = wmma_bf(aq[0][di], bk0, s[0][0]);
      s[0][1] = wmma_bf(aq[0][di], bk1, s[0][1]);
      s[1][0] = wmma_bf(aq[1][di], bk0, s[1][0]);
      s[1][1] = wmma_bf(aq[1][di], bk1, s[1][1]);
    }
    // ---- scale + bias (bias read exactly once, fused) ----
#pragma unroll
    for (int qi = 0; qi < 2; ++qi)
#pragma unroll
      for (int ni = 0; ni < 2; ++ni) {
        const float* brow =
            Bp + (size_t)(q0 + qi * 16 + (hh << 3)) * SEQ + kv0 + ni * 16 + n;
#pragma unroll
        for (int r = 0; r < 8; ++r)
          s[qi][ni][r] = s[qi][ni][r] * 0.125f + brow[(size_t)r * SEQ];
      }
    // ---- online softmax ----
#pragma unroll
    for (int qi = 0; qi < 2; ++qi) {
      float alpha[8];
#pragma unroll
      for (int r = 0; r < 8; ++r) {
        float tmax = fmaxf(s[qi][0][r], s[qi][1][r]);
#pragma unroll
        for (int off = 1; off < 16; off <<= 1)
          tmax = fmaxf(tmax, __shfl_xor(tmax, off, 32));
        const float mn = fmaxf(mrow[qi][r], tmax);
        const float al = __expf(mrow[qi][r] - mn);
        const float p0 = __expf(s[qi][0][r] - mn);
        const float p1 = __expf(s[qi][1][r] - mn);
        s[qi][0][r] = p0;
        s[qi][1][r] = p1;
        float rs = p0 + p1;
#pragma unroll
        for (int off = 1; off < 16; off <<= 1) rs += __shfl_xor(rs, off, 32);
        lrow[qi][r] = lrow[qi][r] * al + rs;
        mrow[qi][r] = mn;
        alpha[r] = al;
      }
#pragma unroll
      for (int nt = 0; nt < 4; ++nt)
#pragma unroll
        for (int r = 0; r < 8; ++r) o[qi][nt][r] *= alpha[r];
      // P -> LDS (C-layout scatter; same-wave DS ops stay in order)
#pragma unroll
      for (int ni = 0; ni < 2; ++ni)
#pragma unroll
        for (int r = 0; r < 8; ++r)
          Pl[wave][qi * 16 + (hh << 3) + r][ni * 16 + n] = f2bf(s[qi][ni][r]);
    }
    // ---- O += P @ V (pre-swizzled V: contiguous 32B fragment loads) ----
    const unsigned short* vt = Vp + (((size_t)(kv0 >> 5) * 4) << 9);
    v16bf bv[4];
#pragma unroll
    for (int nt = 0; nt < 4; ++nt)
      bv[nt] = *(const v16bf*)(vt + ((size_t)nt << 9));
#pragma unroll
    for (int qi = 0; qi < 2; ++qi) {
      const v16bf ap = load_a_lds(&Pl[wave][qi * 16][0], lane);
#pragma unroll
      for (int nt = 0; nt < 4; ++nt) o[qi][nt] = wmma_bf(ap, bv[nt], o[qi][nt]);
    }
  }
  // ---- epilogue: O / l -> attn_out_prelinear (f32 output + bf16 copy) ----
#pragma unroll
  for (int qi = 0; qi < 2; ++qi)
#pragma unroll
    for (int nt = 0; nt < 4; ++nt)
#pragma unroll
      for (int r = 0; r < 8; ++r) {
        const float inv = 1.f / lrow[qi][r];
        const int row = q0 + qi * 16 + (hh << 3) + r;
        const int col = h * DHEAD + nt * 16 + n;
        const size_t idx = ((size_t)b * SEQ + row) * HD + col;
        const float val = o[qi][nt][r] * inv;
        Of[idx] = val;
        Ob[idx] = f2bf(val);
      }
}

// ---------------------------------------------------------------------------
// Elementwise bf16 multiply (bilinear FFN gate): a = a * b
// ---------------------------------------------------------------------------
__global__ __launch_bounds__(256) void mul_kernel(unsigned short* __restrict__ a,
                                                  const unsigned short* __restrict__ b,
                                                  int n) {
  const int i = blockIdx.x * 256 + threadIdx.x;
  if (i < n) a[i] = f2bf(bf2f(a[i]) * bf2f(b[i]));
}

// ---------------------------------------------------------------------------
extern "C" void kernel_launch(void* const* d_in, const int* in_sizes, int n_in,
                              void* d_out, int out_size, void* d_ws, size_t ws_size,
                              hipStream_t stream) {
  (void)in_sizes; (void)n_in; (void)out_size; (void)ws_size;

  const float* x     = (const float*)d_in[0];
  const float* bias  = (const float*)d_in[1];
  const float* W_Q   = (const float*)d_in[2];
  const float* W_K   = (const float*)d_in[3];
  const float* W_V   = (const float*)d_in[4];
  const float* W_O   = (const float*)d_in[5];
  const float* W_1   = (const float*)d_in[6];
  const float* W_2   = (const float*)d_in[7];
  const float* W_3   = (const float*)d_in[8];
  const float* ln1_g = (const float*)d_in[9];
  const float* ln1_b = (const float*)d_in[10];
  const float* ln2_g = (const float*)d_in[11];
  const float* ln2_b = (const float*)d_in[12];
  const float* ln3_g = (const float*)d_in[13];
  const float* ln3_b = (const float*)d_in[14];
  const float* ln4_g = (const float*)d_in[15];
  const float* ln4_b = (const float*)d_in[16];

  float* out_post_attn = (float*)d_out;                       // output 0
  float* out_post_ff   = out_post_attn + (size_t)ROWS * HD;   // output 1
  float* out_attnpre   = out_post_ff   + (size_t)ROWS * HD;   // output 2
  float* out_ffout     = out_attnpre   + (size_t)ROWS * HD;   // output 3

  // ---- workspace carve (live-range reuse, ~136 MB) ----
  char* wp = (char*)d_ws;
  auto carve = [&](size_t bytes) -> char* {
    char* p = wp;
    wp += (bytes + 255) & ~(size_t)255;
    return p;
  };
  const size_t HH = (size_t)HD * HD;        // 1M
  const size_t H4 = (size_t)HD * 4 * HD;    // 4M
  const size_t RH = (size_t)ROWS * HD;      // 4M
  const size_t R4 = (size_t)ROWS * 4 * HD;  // 16M

  unsigned short* wq_bf = (unsigned short*)carve(HH * 2);
  unsigned short* wk_bf = (unsigned short*)carve(HH * 2);
  unsigned short* wv_bf = (unsigned short*)carve(HH * 2);
  unsigned short* wo_bf = (unsigned short*)carve(HH * 2);
  unsigned short* w1_bf = (unsigned short*)carve(H4 * 2);
  unsigned short* w2_bf = (unsigned short*)carve(H4 * 2);
  unsigned short* w3_bf = (unsigned short*)carve(H4 * 2);
  char* bufA = carve(RH * 2 * 3);  // q/k/v  -> later oproj_f(16MB)+xn2_bf(8MB)
  char* bufB = carve(RH * 2);      // xn_bf  -> later apre_bf
  char* bufC = carve(R4 * 2);      // g1
  char* bufD = carve(R4 * 2);      // g2     -> later ff_f
  char* bufE = carve(RH * 2);      // packed V

  unsigned short* q_bf    = (unsigned short*)bufA;
  unsigned short* k_bf    = q_bf + RH;
  unsigned short* v_bf    = k_bf + RH;
  unsigned short* xn_bf   = (unsigned short*)bufB;
  unsigned short* apre_bf = (unsigned short*)bufB;              // reuse after QKV
  float*          oproj_f = (float*)bufA;                       // reuse after attn
  unsigned short* xn2_bf  = (unsigned short*)(bufA + RH * 4);   // past oproj_f
  unsigned short* g1_bf   = (unsigned short*)bufC;
  unsigned short* g2_bf   = (unsigned short*)bufD;
  float*          ff_f    = (float*)bufD;                       // reuse after mul
  unsigned short* vp_bf   = (unsigned short*)bufE;

  // ---- weights -> packed bf16 B-fragments (convert + swizzle in one pass) --
  // ntShift = log2(N/16): 6 for N=1024, 8 for N=4096.
  pack_b_kernel<<<(int)((HH + 255) / 256), 256, 0, stream>>>(W_Q, wq_bf, HD, 6, (int)HH);
  pack_b_kernel<<<(int)((HH + 255) / 256), 256, 0, stream>>>(W_K, wk_bf, HD, 6, (int)HH);
  pack_b_kernel<<<(int)((HH + 255) / 256), 256, 0, stream>>>(W_V, wv_bf, HD, 6, (int)HH);
  pack_b_kernel<<<(int)((HH + 255) / 256), 256, 0, stream>>>(W_O, wo_bf, HD, 6, (int)HH);
  pack_b_kernel<<<(int)((H4 + 255) / 256), 256, 0, stream>>>(W_1, w1_bf, 4 * HD, 8, (int)H4);
  pack_b_kernel<<<(int)((H4 + 255) / 256), 256, 0, stream>>>(W_2, w2_bf, 4 * HD, 8, (int)H4);
  pack_b_kernel<<<(int)((H4 + 255) / 256), 256, 0, stream>>>(W_3, w3_bf, HD, 6, (int)H4);

  const dim3 gblock(32, 4);
  // waves = (M/32)*(N/64); tnShift = log2(N/64): 4 for N=1024, 6 for N=4096.
  auto gemm_grid = [](int M, int N) {
    return dim3((unsigned)(((M / 32) * (N / 64) + 3) / 4));
  };

  // ---- ln1 -> xn (bf16) ----
  ln_kernel<<<ROWS, 256, 0, stream>>>(x, nullptr, ln1_g, ln1_b,
                                      nullptr, nullptr, xn_bf);
  // ---- QKV projections ----
  gemm_bf16_kernel<<<gemm_grid(ROWS, HD), gblock, 0, stream>>>(
      xn_bf, wq_bf, nullptr, q_bf, ROWS, HD, HD, 4);
  gemm_bf16_kernel<<<gemm_grid(ROWS, HD), gblock, 0, stream>>>(
      xn_bf, wk_bf, nullptr, k_bf, ROWS, HD, HD, 4);
  gemm_bf16_kernel<<<gemm_grid(ROWS, HD), gblock, 0, stream>>>(
      xn_bf, wv_bf, nullptr, v_bf, ROWS, HD, HD, 4);
  // ---- V -> WMMA fragment order ----
  pack_v_kernel<<<(int)(RH / 256), 256, 0, stream>>>(v_bf, vp_bf);
  // ---- flash attention (writes attn_out_prelinear f32 + bf16) ----
  attn_kernel<<<dim3((BATCH * NHEADS * (SEQ / 32)) / 4), gblock, 0, stream>>>(
      q_bf, k_bf, vp_bf, bias, out_attnpre, apre_bf);
  // ---- output projection + ln2 + residual ----
  gemm_bf16_kernel<<<gemm_grid(ROWS, HD), gblock, 0, stream>>>(
      apre_bf, wo_bf, oproj_f, nullptr, ROWS, HD, HD, 4);
  ln_kernel<<<ROWS, 256, 0, stream>>>(oproj_f, x, ln2_g, ln2_b,
                                      out_post_attn, nullptr, nullptr);
  // ---- ln3 -> xn2 (bf16) ----
  ln_kernel<<<ROWS, 256, 0, stream>>>(out_post_attn, nullptr, ln3_g, ln3_b,
                                      nullptr, nullptr, xn2_bf);
  // ---- bilinear FFN ----
  gemm_bf16_kernel<<<gemm_grid(ROWS, 4 * HD), gblock, 0, stream>>>(
      xn2_bf, w1_bf, nullptr, g1_bf, ROWS, 4 * HD, HD, 6);
  gemm_bf16_kernel<<<gemm_grid(ROWS, 4 * HD), gblock, 0, stream>>>(
      xn2_bf, w2_bf, nullptr, g2_bf, ROWS, 4 * HD, HD, 6);
  mul_kernel<<<(int)((R4 + 255) / 256), 256, 0, stream>>>(g1_bf, g2_bf, (int)R4);
  gemm_bf16_kernel<<<gemm_grid(ROWS, HD), gblock, 0, stream>>>(
      g1_bf, w3_bf, ff_f, nullptr, ROWS, HD, 4 * HD, 4);
  // ---- ln4 -> ff_out, x_post_ff = x_post_attn + ff_out ----
  ln_kernel<<<ROWS, 256, 0, stream>>>(ff_f, out_post_attn, ln4_g, ln4_b,
                                      out_post_ff, out_ffout, nullptr);
}
